// NearestNeighborAttention_66932770341179
// MI455X (gfx1250) — matrix operand order repaired and checked by reading, not compiled, not run on hardware
//
#include <hip/hip_runtime.h>
#include <hip/hip_bf16.h>

typedef __attribute__((ext_vector_type(16))) _Float16 v16h;
typedef __attribute__((ext_vector_type(8)))  _Float16 v8h;
typedef __attribute__((ext_vector_type(8)))  float    v8f;

#define B_  4
#define S_  2048
#define F_  512
#define H_  8
#define D_  64
#define K_  32
#define BS_ (B_ * S_)                       // 8192 rows
static constexpr size_t NX  = (size_t)BS_ * F_;   // 4,194,304 elems of x
static constexpr size_t NW  = (size_t)F_ * F_;    // 262,144 elems per weight

#define LDS_STRIDE 264   // 256 K-halves + 8 pad halves -> 528B: 4-bank rotate/col

// ---------------------------------------------------------------------------
// 1) f32 -> f16 conversion of x and the three weight matrices
// ---------------------------------------------------------------------------
__global__ void __launch_bounds__(256) convert_kernel(
    const float* __restrict__ x,
    const float* __restrict__ wq, const float* __restrict__ wk,
    const float* __restrict__ wv,
    _Float16* __restrict__ xh, _Float16* __restrict__ wh) {
  size_t i = (size_t)blockIdx.x * blockDim.x + threadIdx.x;
  if (i >= NX + 3 * NW) return;
  if (i < NX) { xh[i] = (_Float16)x[i]; return; }
  size_t j = i - NX;
  const float* w = (j < NW) ? wq : (j < 2 * NW) ? wk : wv;
  size_t r = (j < NW) ? j : (j < 2 * NW) ? (j - NW) : (j - 2 * NW);
  wh[j] = (_Float16)w[r];
}

// ---------------------------------------------------------------------------
// 2) kNN: one thread per (b,q); unordered top-32 smallest squared distance
// ---------------------------------------------------------------------------
__global__ void __launch_bounds__(256) knn_kernel(
    const float* __restrict__ coords, int* __restrict__ nn_idx) {
  int gid = blockIdx.x * blockDim.x + threadIdx.x;
  if (gid >= BS_) return;
  int b = gid / S_, q = gid % S_;
  const float* cb = coords + (size_t)b * S_ * 3;
  float qx = cb[q * 3 + 0], qy = cb[q * 3 + 1], qz = cb[q * 3 + 2];
  float bd[K_]; int bi[K_];
#pragma unroll
  for (int n = 0; n < K_; ++n) { bd[n] = 3.0e38f; bi[n] = 0; }
  float worst = 3.0e38f; int wpos = 0;
  for (int j = 0; j < S_; ++j) {
    if (j == q) continue;
    float dx = qx - cb[j * 3 + 0];
    float dy = qy - cb[j * 3 + 1];
    float dz = qz - cb[j * 3 + 2];
    float d2 = dx * dx + dy * dy + dz * dz;
    if (d2 < worst) {
      bd[wpos] = d2; bi[wpos] = j;
      worst = bd[0]; wpos = 0;
      for (int n = 1; n < K_; ++n)
        if (bd[n] > worst) { worst = bd[n]; wpos = n; }
    }
  }
  int* o = nn_idx + (size_t)gid * K_;
  for (int n = 0; n < K_; ++n) o[n] = bi[n];
}

// ---------------------------------------------------------------------------
// 3) QKV projections via v_wmma_f32_16x16x32_f16.
//    Block = (mat, nblk, 8 mtiles). Weight strip staged in LDS with
//    global_load_async_to_lds_b128 (two 256-K phases); 8 waves share it.
//    A frags and B frags are both register-double-buffered so ds/global
//    latency overlaps the WMMA issue.
// ---------------------------------------------------------------------------
#define LOADB(dst, ks, nn)                                                     \
  {                                                                            \
    const _Float16* bp_ = bbase + (ks) * 32 + hi * 8 + (nn) * 16 * LDS_STRIDE;\
    v8h t0_ = *(const v8h*)(bp_);                                              \
    v8h t1_ = *(const v8h*)(bp_ + 16);                                         \
    dst = __builtin_shufflevector(t0_, t1_,                                    \
        0, 1, 2, 3, 4, 5, 6, 7, 8, 9, 10, 11, 12, 13, 14, 15);                 \
  }

__global__ void __launch_bounds__(256) qkv_gemm_kernel(
    const _Float16* __restrict__ Xh, const _Float16* __restrict__ Wh,
    float* __restrict__ Qp, float* __restrict__ Kp, float* __restrict__ Vp) {
  __shared__ _Float16 ldsB[64 * LDS_STRIDE];   // 33,792 B

  int tid  = threadIdx.x;
  int lane = tid & 31;
  int wv   = tid >> 5;                 // wave in block: 0..7
  int bx   = blockIdx.x;               // 0..1535
  int mat  = bx / 512;                 // 3 matrices
  int rem  = bx % 512;
  int mgrp = rem >> 3;                 // 64 groups of 8 mtiles
  int nblk = rem & 7;                  // 8 strips of 64 cols
  int mtile = mgrp * 8 + wv;
  int lrow = lane & 15;
  int hi   = lane >> 4;

  const _Float16* a_base = Xh + (size_t)(mtile * 16 + lrow) * F_;
  const _Float16* w_strip = Wh + (size_t)mat * NW + (size_t)(nblk * 64) * F_;
  unsigned lds_base = (unsigned)(uintptr_t)(&ldsB[0]);

  v8f acc0 = {}, acc1 = {}, acc2 = {}, acc3 = {};

  for (int p = 0; p < 2; ++p) {
    // ---- async stage: 64 cols x 256 K halves (32 KB) into LDS ----
#pragma unroll
    for (int i = 0; i < 8; ++i) {
      int ch  = i * 256 + tid;                 // 2048 chunks of 8 halves
      int col = ch >> 5;                       // 0..63
      int ko  = (ch & 31) * 8;                 // 0..248 halves
      const _Float16* g = w_strip + (size_t)col * F_ + p * 256 + ko;
      unsigned l = lds_base + (unsigned)(col * LDS_STRIDE + ko) * 2;
      asm volatile("global_load_async_to_lds_b128 %0, %1, off"
                   :: "v"(l), "v"(g) : "memory");
    }
    asm volatile("s_wait_asynccnt 0x0" ::: "memory");
    __syncthreads();

    // ---- compute 8 k-steps; A and B frags double-buffered ----
    const _Float16* bbase = &ldsB[lrow * LDS_STRIDE];
    const _Float16* ap0 = a_base + p * 256 + hi * 8;
    v8h alo = *(const v8h*)(ap0);
    v8h ahi = *(const v8h*)(ap0 + 16);
    v16h b0, b1, b2, b3;
    LOADB(b0, 0, 0); LOADB(b1, 0, 1); LOADB(b2, 0, 2); LOADB(b3, 0, 3);
#pragma unroll
    for (int ks = 0; ks < 8; ++ks) {
      v16h a = __builtin_shufflevector(alo, ahi,
          0, 1, 2, 3, 4, 5, 6, 7, 8, 9, 10, 11, 12, 13, 14, 15);
      v16h n0, n1, n2, n3;
      if (ks < 7) {
        // prefetch next k-step's A (global) and B (LDS) before issuing WMMAs
        const _Float16* apn = a_base + p * 256 + (ks + 1) * 32 + hi * 8;
        alo = *(const v8h*)(apn);
        ahi = *(const v8h*)(apn + 16);
        LOADB(n0, ks + 1, 0); LOADB(n1, ks + 1, 1);
        LOADB(n2, ks + 1, 2); LOADB(n3, ks + 1, 3);
      }
      acc0 = __builtin_amdgcn_wmma_f32_16x16x32_f16(false, a, false, b0, (short)0, acc0, false, false);
      acc1 = __builtin_amdgcn_wmma_f32_16x16x32_f16(false, a, false, b1, (short)0, acc1, false, false);
      acc2 = __builtin_amdgcn_wmma_f32_16x16x32_f16(false, a, false, b2, (short)0, acc2, false, false);
      acc3 = __builtin_amdgcn_wmma_f32_16x16x32_f16(false, a, false, b3, (short)0, acc3, false, false);
      if (ks < 7) { b0 = n0; b1 = n1; b2 = n2; b3 = n3; }
    }
    __syncthreads();   // LDS reuse barrier before next phase's staging
  }

  float* Y = (mat == 0) ? Qp : (mat == 1) ? Kp : Vp;
  int colbase = nblk * 64 + lrow;
  int rowbase = mtile * 16 + hi * 8;    // C/D: row = v + (lane>>4)*8, col = lane&15
#pragma unroll
  for (int v = 0; v < 8; ++v) {
    size_t r = (size_t)(rowbase + v) * F_;
    Y[r + colbase +  0] = acc0[v];
    Y[r + colbase + 16] = acc1[v];
    Y[r + colbase + 32] = acc2[v];
    Y[r + colbase + 48] = acc3[v];
  }
}

// ---------------------------------------------------------------------------
// 4) Sparse neighbor attention: one wave32 per (b,h,q)
// ---------------------------------------------------------------------------
__global__ void __launch_bounds__(256) attn_kernel(
    const float* __restrict__ Qp, const float* __restrict__ Kp,
    const float* __restrict__ Vp, const int* __restrict__ nn_idx,
    float* __restrict__ out) {
  int tid  = threadIdx.x;
  int lane = tid & 31;
  int wave = (int)((blockIdx.x * blockDim.x + tid) >> 5); // 0..65535
  int b = wave >> 14;          // H_*S_ = 16384
  int h = (wave >> 11) & 7;
  int q = wave & (S_ - 1);

  size_t qrow = ((size_t)b * S_ + q) * F_ + (size_t)h * D_;
  float q0 = Qp[qrow + lane];
  float q1 = Qp[qrow + 32 + lane];
  int nbr = nn_idx[((size_t)b * S_ + q) * K_ + lane];

  float myscore = 0.0f;
  for (int n = 0; n < K_; ++n) {
    int kj = __shfl(nbr, n, 32);
    size_t krow = ((size_t)b * S_ + kj) * F_ + (size_t)h * D_;
    float p = q0 * Kp[krow + lane] + q1 * Kp[krow + 32 + lane];
    p += __shfl_xor(p, 16, 32);
    p += __shfl_xor(p, 8, 32);
    p += __shfl_xor(p, 4, 32);
    p += __shfl_xor(p, 2, 32);
    p += __shfl_xor(p, 1, 32);
    if (lane == n) myscore = p * 0.125f;   // 1/sqrt(64)
  }
  float m = myscore;
  m = fmaxf(m, __shfl_xor(m, 16, 32));
  m = fmaxf(m, __shfl_xor(m, 8, 32));
  m = fmaxf(m, __shfl_xor(m, 4, 32));
  m = fmaxf(m, __shfl_xor(m, 2, 32));
  m = fmaxf(m, __shfl_xor(m, 1, 32));
  float e = __expf(myscore - m);
  float s = e;
  s += __shfl_xor(s, 16, 32);
  s += __shfl_xor(s, 8, 32);
  s += __shfl_xor(s, 4, 32);
  s += __shfl_xor(s, 2, 32);
  s += __shfl_xor(s, 1, 32);
  float a = e / s;
  float o0 = 0.0f, o1 = 0.0f;
  for (int n = 0; n < K_; ++n) {
    float an = __shfl(a, n, 32);
    int   vj = __shfl(nbr, n, 32);
    size_t vrow = ((size_t)b * S_ + vj) * F_ + (size_t)h * D_;
    o0 += an * Vp[vrow + lane];
    o1 += an * Vp[vrow + 32 + lane];
  }
  out[qrow + lane]      = o0;
  out[qrow + 32 + lane] = o1;
}

// ---------------------------------------------------------------------------
// 5) metric[b,h,d] = mean over s of Kp[b,s,h*64+d]
// ---------------------------------------------------------------------------
__global__ void __launch_bounds__(256) metric_kernel(
    const float* __restrict__ Kp, float* __restrict__ metric) {
  int b = blockIdx.x;
  int f = blockIdx.y * blockDim.x + threadIdx.x;   // 0..511
  const float* base = Kp + (size_t)b * S_ * F_ + f;
  float s = 0.0f;
  for (int j = 0; j < S_; ++j) s += base[(size_t)j * F_];
  metric[b * F_ + f] = s * (1.0f / S_);
}

// ---------------------------------------------------------------------------
extern "C" void kernel_launch(void* const* d_in, const int* in_sizes, int n_in,
                              void* d_out, int out_size, void* d_ws, size_t ws_size,
                              hipStream_t stream) {
  (void)in_sizes; (void)n_in; (void)out_size; (void)ws_size;
  const float* x      = (const float*)d_in[0];
  const float* coords = (const float*)d_in[1];
  const float* Wq     = (const float*)d_in[2];
  const float* Wk     = (const float*)d_in[3];
  const float* Wv     = (const float*)d_in[4];
  float* out = (float*)d_out;

  char* ws = (char*)d_ws;
  constexpr size_t XH_BYTES = NX * 2;            // 8 MB
  constexpr size_t WH_BYTES = 3 * NW * 2;        // 1.5 MB
  constexpr size_t YP_BYTES = NX * 4;            // 16 MB each
  _Float16* Xh = (_Float16*)ws;
  _Float16* Wh = (_Float16*)(ws + XH_BYTES);
  float* Qp = (float*)(ws + XH_BYTES + WH_BYTES);
  float* Kp = (float*)(ws + XH_BYTES + WH_BYTES + YP_BYTES);
  float* Vp = (float*)(ws + XH_BYTES + WH_BYTES + 2 * YP_BYTES);
  int*   nn = (int*)  (ws + XH_BYTES + WH_BYTES + 3 * YP_BYTES);

  {
    size_t total = NX + 3 * NW;                  // 4,980,736
    int blocks = (int)((total + 255) / 256);
    convert_kernel<<<blocks, 256, 0, stream>>>(x, Wq, Wk, Wv, Xh, Wh);
  }
  knn_kernel<<<BS_ / 256, 256, 0, stream>>>(coords, nn);
  // 3 mats * 64 mgroups * 8 nblks = 1536 blocks of 8 waves
  qkv_gemm_kernel<<<1536, 256, 0, stream>>>(Xh, Wh, Qp, Kp, Vp);
  attn_kernel<<<8192, 256, 0, stream>>>(Qp, Kp, Vp, nn, out);
  metric_kernel<<<dim3(B_, 2), 256, 0, stream>>>(Kp, out + NX);
}
